// RSSMRollout_10402410790926
// MI455X (gfx1250) — matrix operand order repaired and checked by reading, not compiled, not run on hardware
//
#include <hip/hip_runtime.h>

typedef __attribute__((ext_vector_type(16))) __bf16 v16bf;
typedef __attribute__((ext_vector_type(8)))  __bf16 v8bf;
typedef __attribute__((ext_vector_type(8)))  float  v8f;

#define WMMA_BF16(a, b, c) \
    __builtin_amdgcn_wmma_f32_16x16x32_bf16(false, (a), false, (b), (short)0, (c), false, false)

// ---------------------------------------------------------------------------
// Weight prep: fp32 -> bf16, swizzled into WMMA B-fragment order.
// Layout: tile index = nt*nkt + kt ; within tile: lane*16 + h  (512 bf16/tile)
// B-operand mapping (32x16, K x N): n = lane&15 ; k = kt*32 + ((h&8)<<1) + ((lane>>4)<<3) + (h&7)
// transpose==1: source is [N][K] row-major (W_ih/W_hh, used as x @ W.T)
// ---------------------------------------------------------------------------
__global__ void prep_weight(const float* __restrict__ W, __bf16* __restrict__ out,
                            int nkt, int Ksrc, int stride, int transpose, int total) {
    int idx = blockIdx.x * 256 + threadIdx.x;
    if (idx >= total) return;
    int h    = idx & 15;
    int lane = (idx >> 4) & 31;
    int tile = idx >> 9;
    int kt   = tile % nkt;
    int nt   = tile / nkt;
    int k = kt * 32 + ((h & 8) << 1) + ((lane >> 4) << 3) + (h & 7);
    int n = nt * 16 + (lane & 15);
    float v = 0.f;
    if (k < Ksrc) v = transpose ? W[n * stride + k] : W[k * stride + n];
    out[idx] = (__bf16)v;
}

// ---------------------------------------------------------------------------
// Fragment loaders
// ---------------------------------------------------------------------------
__device__ __forceinline__ v16bf ldsA(const __bf16* base, int ld, int k0, int lane) {
    int m  = lane & 15;
    int ko = (lane >> 4) << 3;
    const __bf16* p = base + m * ld + k0 + ko;
    v8bf lo = *(const v8bf*)(p);
    v8bf hi = *(const v8bf*)(p + 16);
    v16bf r;
#pragma unroll
    for (int i = 0; i < 8; ++i) { r[i] = lo[i]; r[i + 8] = hi[i]; }
    return r;
}

__device__ __forceinline__ v16bf ldB(const __bf16* tileBase, int lane) {
    const __bf16* p = tileBase + lane * 16;
    v8bf lo = *(const v8bf*)(p);
    v8bf hi = *(const v8bf*)(p + 8);
    v16bf r;
#pragma unroll
    for (int i = 0; i < 8; ++i) { r[i] = lo[i]; r[i + 8] = hi[i]; }
    return r;
}

__device__ __forceinline__ float elu_f(float x)      { return x > 0.f ? x : (__expf(x) - 1.f); }
__device__ __forceinline__ float sigmoid_f(float x)  { return 1.f / (1.f + __expf(-x)); }
__device__ __forceinline__ float softplus_f(float x) { return x > 20.f ? x : log1pf(__expf(x)); }

// ---------------------------------------------------------------------------
// Persistent rollout: one workgroup = 16 batch rows, iterates all T=64 steps.
// ---------------------------------------------------------------------------
__global__ __launch_bounds__(256) void rssm_rollout(
    const float* __restrict__ post0, const float* __restrict__ bel0,
    const float* __restrict__ actions, const float* __restrict__ obs,
    const float* __restrict__ eps_p, const float* __restrict__ eps_q,
    const float* __restrict__ b_as, const float* __restrict__ b_ih,
    const float* __restrict__ b_hh, const float* __restrict__ b_p1,
    const float* __restrict__ b_p2, const float* __restrict__ b_q1,
    const float* __restrict__ b_q2,
    const __bf16* __restrict__ w_as, const __bf16* __restrict__ w_ih,
    const __bf16* __restrict__ w_hh, const __bf16* __restrict__ w_p1,
    const __bf16* __restrict__ w_p2, const __bf16* __restrict__ w_q1,
    const __bf16* __restrict__ w_q2,
    float* __restrict__ out_pmean, float* __restrict__ out_pstd,
    float* __restrict__ out_prior, float* __restrict__ out_bel,
    float* __restrict__ out_qmean, float* __restrict__ out_qstd,
    float* __restrict__ out_post) {

    __shared__ float  s_bel[16 * 512];    // fp32 recurrent belief          32 KB
    __shared__ __bf16 s_belb[16 * 512];   // bf16 copy (GEMM A operand)     16 KB
    __shared__ float  s_post[16 * 64];    // fp32 recurrent posterior        4 KB
    __shared__ __bf16 s_ac[16 * 96];      // cat(post, action), K-padded     3 KB
    __shared__ __bf16 s_x[16 * 512];      // x / h_p / h_q staging          16 KB
    __shared__ __bf16 s_obs[16 * 1024];   // obs tile (bf16)                32 KB
    __shared__ float  s_pq[16 * 128];     // mean/std staging                8 KB

    const int tid  = threadIdx.x;
    const int lane = tid & 31;
    const int w    = tid >> 5;
    const int b0   = blockIdx.x * 16;

    for (int i = tid; i < 16 * 512; i += 256) {
        int m = i >> 9, j = i & 511;
        float v = bel0[(b0 + m) * 512 + j];
        s_bel[i]  = v;
        s_belb[i] = (__bf16)v;
    }
    for (int i = tid; i < 16 * 64; i += 256) {
        int m = i >> 6, j = i & 63;
        s_post[i] = post0[(b0 + m) * 64 + j];
    }
    __syncthreads();

    for (int t = 0; t < 64; ++t) {
        // ---- stage cat(post_s, action) (K padded 80->96) ----
        for (int i = tid; i < 16 * 96; i += 256) {
            int m = i / 96, c = i % 96;
            float v = 0.f;
            if (c < 64)      v = s_post[m * 64 + c];
            else if (c < 80) v = actions[((b0 + m) * 64 + t) * 16 + (c - 64)];
            s_ac[i] = (__bf16)v;
        }
        // ---- stage obs tile, vectorized f32x4 -> bf16x4 ----
        for (int i = tid; i < 16 * 256; i += 256) {           // 4 elems per iter
            int m = i >> 8, e4 = (i & 255) << 2;
            const float4 v = *(const float4*)(obs + ((b0 + m) * 64 + t) * 1024 + e4);
            __bf16* d = s_obs + m * 1024 + e4;
            d[0] = (__bf16)v.x; d[1] = (__bf16)v.y; d[2] = (__bf16)v.z; d[3] = (__bf16)v.w;
        }
        __syncthreads();

        // ---- GEMM1: x = elu([post,a] @ W_as + b_as)   [16,96]x[96,512] ----
#pragma unroll 1
        for (int nt = w * 4; nt < w * 4 + 4; ++nt) {
            v8f acc = {};
#pragma unroll
            for (int kt = 0; kt < 3; ++kt) {
                v16bf af = ldsA(s_ac, 96, kt * 32, lane);
                v16bf bf = ldB(w_as + ((nt * 3 + kt) << 9), lane);
                acc = WMMA_BF16(af, bf, acc);
            }
            int n  = nt * 16 + (lane & 15);
            int mb = (lane >> 4) << 3;
            float bias = b_as[n];
#pragma unroll
            for (int vi = 0; vi < 8; ++vi)
                s_x[(vi + mb) * 512 + n] = (__bf16)elu_f(acc[vi] + bias);
        }
        __syncthreads();

        // ---- GRU gates: 4 accumulators (r,z fused gi+gh; inn/hn separate) ----
#pragma unroll 1
        for (int jt = w * 4; jt < w * 4 + 4; ++jt) {
            v8f ar = {}, az = {}, ai = {}, ah = {};
#pragma unroll 4
            for (int kt = 0; kt < 16; ++kt) {
                v16bf ax = ldsA(s_x, 512, kt * 32, lane);
                v16bf ab = ldsA(s_belb, 512, kt * 32, lane);
                v16bf bf;
                bf = ldB(w_ih + (((jt     ) * 16 + kt) << 9), lane); ar = WMMA_BF16(ax, bf, ar);
                bf = ldB(w_hh + (((jt     ) * 16 + kt) << 9), lane); ar = WMMA_BF16(ab, bf, ar);
                bf = ldB(w_ih + (((jt + 32) * 16 + kt) << 9), lane); az = WMMA_BF16(ax, bf, az);
                bf = ldB(w_hh + (((jt + 32) * 16 + kt) << 9), lane); az = WMMA_BF16(ab, bf, az);
                bf = ldB(w_ih + (((jt + 64) * 16 + kt) << 9), lane); ai = WMMA_BF16(ax, bf, ai);
                bf = ldB(w_hh + (((jt + 64) * 16 + kt) << 9), lane); ah = WMMA_BF16(ab, bf, ah);
            }
            int j  = jt * 16 + (lane & 15);
            int mb = (lane >> 4) << 3;
            float brz = b_ih[j] + b_hh[j];
            float bzz = b_ih[512 + j] + b_hh[512 + j];
            float bin = b_ih[1024 + j];
            float bhn = b_hh[1024 + j];
#pragma unroll
            for (int vi = 0; vi < 8; ++vi) {
                int m = vi + mb;
                float r  = sigmoid_f(ar[vi] + brz);
                float z  = sigmoid_f(az[vi] + bzz);
                float ng = tanhf(ai[vi] + bin + r * (ah[vi] + bhn));
                float bo = s_bel[m * 512 + j];
                float bn = (1.f - z) * ng + z * bo;
                s_bel[m * 512 + j] = bn;
                out_bel[((b0 + m) * 64 + t) * 512 + j] = bn;
            }
        }
        __syncthreads();
        // refresh bf16 belief (vectorized x4)
        for (int i = tid; i < 16 * 128; i += 256) {
            int i4 = i << 2;
            const float4 v = *(const float4*)(s_bel + i4);
            __bf16* d = s_belb + i4;
            d[0] = (__bf16)v.x; d[1] = (__bf16)v.y; d[2] = (__bf16)v.z; d[3] = (__bf16)v.w;
        }
        __syncthreads();

        // ---- p1: h_p = elu(bel_new @ W_p1 + b_p1) ----
#pragma unroll 1
        for (int nt = w * 4; nt < w * 4 + 4; ++nt) {
            v8f acc = {};
#pragma unroll 4
            for (int kt = 0; kt < 16; ++kt) {
                v16bf af = ldsA(s_belb, 512, kt * 32, lane);
                v16bf bf = ldB(w_p1 + ((nt * 16 + kt) << 9), lane);
                acc = WMMA_BF16(af, bf, acc);
            }
            int n  = nt * 16 + (lane & 15);
            int mb = (lane >> 4) << 3;
            float bias = b_p1[n];
#pragma unroll
            for (int vi = 0; vi < 8; ++vi)
                s_x[(vi + mb) * 512 + n] = (__bf16)elu_f(acc[vi] + bias);
        }
        __syncthreads();

        // ---- p2: [16,512]x[512,128] -> mean/std ----
        {
            int nt = w;
            v8f acc = {};
#pragma unroll 4
            for (int kt = 0; kt < 16; ++kt) {
                v16bf af = ldsA(s_x, 512, kt * 32, lane);
                v16bf bf = ldB(w_p2 + ((nt * 16 + kt) << 9), lane);
                acc = WMMA_BF16(af, bf, acc);
            }
            int n  = nt * 16 + (lane & 15);
            int mb = (lane >> 4) << 3;
            float bias = b_p2[n];
#pragma unroll
            for (int vi = 0; vi < 8; ++vi) {
                int m = vi + mb;
                float val = acc[vi] + bias;
                if (n < 64) {
                    s_pq[m * 128 + n] = val;
                    out_pmean[((b0 + m) * 64 + t) * 64 + n] = val;
                } else {
                    float sd = softplus_f(val) + 0.1f;
                    s_pq[m * 128 + n] = sd;
                    out_pstd[((b0 + m) * 64 + t) * 64 + (n - 64)] = sd;
                }
            }
        }
        __syncthreads();
        for (int i = tid; i < 16 * 64; i += 256) {
            int m = i >> 6, c = i & 63;
            float pv = s_pq[m * 128 + c] +
                       s_pq[m * 128 + 64 + c] * eps_p[((b0 + m) * 64 + t) * 64 + c];
            out_prior[((b0 + m) * 64 + t) * 64 + c] = pv;
        }

        // ---- q1: h_q = elu(cat(bel_new, obs) @ W_q1 + b_q1), K=1536 ----
#pragma unroll 1
        for (int nt = w * 4; nt < w * 4 + 4; ++nt) {
            v8f acc = {};
#pragma unroll 4
            for (int kt = 0; kt < 16; ++kt) {          // belief half of K
                v16bf af = ldsA(s_belb, 512, kt * 32, lane);
                v16bf bf = ldB(w_q1 + ((nt * 48 + kt) << 9), lane);
                acc = WMMA_BF16(af, bf, acc);
            }
#pragma unroll 4
            for (int kt = 16; kt < 48; ++kt) {         // obs half of K
                v16bf af = ldsA(s_obs, 1024, kt * 32 - 512, lane);
                v16bf bf = ldB(w_q1 + ((nt * 48 + kt) << 9), lane);
                acc = WMMA_BF16(af, bf, acc);
            }
            int n  = nt * 16 + (lane & 15);
            int mb = (lane >> 4) << 3;
            float bias = b_q1[n];
#pragma unroll
            for (int vi = 0; vi < 8; ++vi)
                s_x[(vi + mb) * 512 + n] = (__bf16)elu_f(acc[vi] + bias);
        }
        __syncthreads();

        // ---- q2 -> q_mean/q_std/post_new ----
        {
            int nt = w;
            v8f acc = {};
#pragma unroll 4
            for (int kt = 0; kt < 16; ++kt) {
                v16bf af = ldsA(s_x, 512, kt * 32, lane);
                v16bf bf = ldB(w_q2 + ((nt * 16 + kt) << 9), lane);
                acc = WMMA_BF16(af, bf, acc);
            }
            int n  = nt * 16 + (lane & 15);
            int mb = (lane >> 4) << 3;
            float bias = b_q2[n];
#pragma unroll
            for (int vi = 0; vi < 8; ++vi) {
                int m = vi + mb;
                float val = acc[vi] + bias;
                if (n < 64) {
                    s_pq[m * 128 + n] = val;
                    out_qmean[((b0 + m) * 64 + t) * 64 + n] = val;
                } else {
                    float sd = softplus_f(val) + 0.1f;
                    s_pq[m * 128 + n] = sd;
                    out_qstd[((b0 + m) * 64 + t) * 64 + (n - 64)] = sd;
                }
            }
        }
        __syncthreads();
        for (int i = tid; i < 16 * 64; i += 256) {
            int m = i >> 6, c = i & 63;
            float pv = s_pq[m * 128 + c] +
                       s_pq[m * 128 + 64 + c] * eps_q[((b0 + m) * 64 + t) * 64 + c];
            out_post[((b0 + m) * 64 + t) * 64 + c] = pv;
            s_post[i] = pv;
        }
        __syncthreads();
    }
}

// ---------------------------------------------------------------------------
extern "C" void kernel_launch(void* const* d_in, const int* in_sizes, int n_in,
                              void* d_out, int out_size, void* d_ws, size_t ws_size,
                              hipStream_t stream) {
    __bf16* ws   = (__bf16*)d_ws;
    __bf16* w_as = ws;
    __bf16* w_ih = w_as + 96 * 512;
    __bf16* w_hh = w_ih + 512 * 1536;
    __bf16* w_p1 = w_hh + 512 * 1536;
    __bf16* w_p2 = w_p1 + 512 * 512;
    __bf16* w_q1 = w_p2 + 512 * 128;
    __bf16* w_q2 = w_q1 + 1536 * 512;

    auto prep = [&](const void* W, __bf16* out, int nkt, int Ndim, int Ksrc,
                    int stride, int transpose) {
        int total = nkt * 32 * Ndim;
        prep_weight<<<(total + 255) / 256, 256, 0, stream>>>(
            (const float*)W, out, nkt, Ksrc, stride, transpose, total);
    };
    prep(d_in[6],  w_as, 3,  512,  80,   512, 0);  // W_as [80,512], K-pad to 96
    prep(d_in[8],  w_ih, 16, 1536, 512,  512, 1);  // W_ih [1536,512] used as x@W.T
    prep(d_in[10], w_hh, 16, 1536, 512,  512, 1);  // W_hh [1536,512] used as h@W.T
    prep(d_in[12], w_p1, 16, 512,  512,  512, 0);  // W_p1 [512,512]
    prep(d_in[14], w_p2, 16, 128,  512,  128, 0);  // W_p2 [512,128]
    prep(d_in[16], w_q1, 48, 512,  1536, 512, 0);  // W_q1 [1536,512]
    prep(d_in[18], w_q2, 16, 128,  512,  128, 0);  // W_q2 [512,128]

    float* out = (float*)d_out;
    const long SZ = 2048L * 64 * 64;          // [B,T,S]
    float* out_pmean = out;
    float* out_pstd  = out + SZ;
    float* out_prior = out + 2 * SZ;
    float* out_bel   = out + 3 * SZ;          // [B,T,R]
    float* out_qmean = out_bel + 2048L * 64 * 512;
    float* out_qstd  = out_qmean + SZ;
    float* out_post  = out_qstd + SZ;

    rssm_rollout<<<128, 256, 0, stream>>>(
        (const float*)d_in[0], (const float*)d_in[1], (const float*)d_in[2],
        (const float*)d_in[3], (const float*)d_in[4], (const float*)d_in[5],
        (const float*)d_in[7], (const float*)d_in[9], (const float*)d_in[11],
        (const float*)d_in[13], (const float*)d_in[15], (const float*)d_in[17],
        (const float*)d_in[19],
        w_as, w_ih, w_hh, w_p1, w_p2, w_q1, w_q2,
        out_pmean, out_pstd, out_prior, out_bel, out_qmean, out_qstd, out_post);
}
